// SimpleGCNModel_6468220747863
// MI455X (gfx1250) — matrix operand order repaired and checked by reading, not compiled, not run on hardware
//
#include <hip/hip_runtime.h>
#include <hip/hip_bf16.h>

typedef __attribute__((ext_vector_type(16))) _Float16 v16h;
typedef __attribute__((ext_vector_type(8)))  float    v8f;

#define N_NODES 100000
#define N_EDGES 1600000
#define N_FEAT  128
#define HIDDEN  64
#define N_OUT   10
#define N_GRAPHS 64

// ---------------- degree / norm prep ----------------

__global__ void k_deg_init(float* deg, int n) {
    int i = blockIdx.x * blockDim.x + threadIdx.x;
    if (i < n) deg[i] = 1.0f;  // self loop
}

__global__ void k_deg_accum(const int* __restrict__ col, float* deg, int e) {
    int i = blockIdx.x * blockDim.x + threadIdx.x;
    if (i < e) atomicAdd(&deg[col[i]], 1.0f);
}

__global__ void k_rsqrt(float* deg, int n) {   // in place: deg -> dis
    int i = blockIdx.x * blockDim.x + threadIdx.x;
    if (i < n) deg[i] = rsqrtf(deg[i]);
}

__global__ void k_norm(const int* __restrict__ row, const int* __restrict__ col,
                       const float* __restrict__ dis, float* __restrict__ norm, int e) {
    int i = blockIdx.x * blockDim.x + threadIdx.x;
    if (i < e) norm[i] = dis[row[i]] * dis[col[i]];
}

// ---------------- weight transpose + f16 convert:  W[K][C] -> Wt[C][K] ----------------

__global__ void k_wt(const float* __restrict__ W, _Float16* __restrict__ Wt, int K, int C) {
    int t = blockIdx.x * blockDim.x + threadIdx.x;
    if (t >= K * C) return;
    int k = t / C, n = t % C;
    Wt[(size_t)n * K + k] = (_Float16)W[t];
}

// ---------------- WMMA GEMM:  T[N][64] = A[N][K] (f32, cvt->f16) @ Wt^T (f16) ----------------
// block = 256 threads = 8 waves; wave -> (row_tile 0..1, col_tile 0..3)
// grid.x = nrows/32 (nrows % 32 == 0)

__global__ void k_gemm_wmma(const float* __restrict__ A, int K,
                            const _Float16* __restrict__ Wt,
                            float* __restrict__ T) {
    const int wave = threadIdx.x >> 5;
    const int lane = threadIdx.x & 31;
    const int rt = wave >> 2;         // 0..1
    const int ct = wave & 3;          // 0..3
    const int mbase = (blockIdx.x * 2 + rt) * 16;
    const int nbase = ct * 16;
    const int lm = lane & 15;
    const int hi = lane >> 4;         // lane half

    const float*    arow = A  + (size_t)(mbase + lm) * K;
    const _Float16* brow = Wt + (size_t)(nbase + lm) * K;

    v8f acc = {};
    for (int k0 = 0; k0 < K; k0 += 32) {
        // A fragment: lane half selects K offset {0,8}; elems 0..7 -> K=ka..ka+7, 8..15 -> K=ka+16..ka+23
        const int ka = k0 + hi * 8;
        const float4 f0 = *(const float4*)(arow + ka);
        const float4 f1 = *(const float4*)(arow + ka + 4);
        const float4 f2 = *(const float4*)(arow + ka + 16);
        const float4 f3 = *(const float4*)(arow + ka + 20);
        v16h a;
        a[0]=(_Float16)f0.x; a[1]=(_Float16)f0.y; a[2]=(_Float16)f0.z; a[3]=(_Float16)f0.w;
        a[4]=(_Float16)f1.x; a[5]=(_Float16)f1.y; a[6]=(_Float16)f1.z; a[7]=(_Float16)f1.w;
        a[8]=(_Float16)f2.x; a[9]=(_Float16)f2.y; a[10]=(_Float16)f2.z; a[11]=(_Float16)f2.w;
        a[12]=(_Float16)f3.x; a[13]=(_Float16)f3.y; a[14]=(_Float16)f3.z; a[15]=(_Float16)f3.w;
        // B fragment: lane = column n, lane half selects K offset {0,8}; elems 0..7 -> K=kb..kb+7, 8..15 -> K=kb+16..kb+23
        const int kb = k0 + hi * 8;
        typedef __attribute__((ext_vector_type(8))) _Float16 v8h_t;
        const v8h_t b0 = *(const v8h_t*)(brow + kb);
        const v8h_t b1 = *(const v8h_t*)(brow + kb + 16);
        v16h b = __builtin_shufflevector(b0, b1, 0,1,2,3,4,5,6,7,8,9,10,11,12,13,14,15);
        acc = __builtin_amdgcn_wmma_f32_16x16x32_f16(
            /*neg_a=*/false, a, /*neg_b=*/false, b,
            /*c_mod=*/(short)0, acc, /*reuse_a=*/false, /*reuse_b=*/false);
    }
    // D layout: VGPR r -> M = mbase + r + hi*8, N = nbase + lm
    #pragma unroll
    for (int r = 0; r < 8; ++r)
        T[(size_t)(mbase + r + hi * 8) * HIDDEN + nbase + lm] = acc[r];
}

// ---------------- aggregation ----------------

// out[i][c] = T[i][c] * dis[i]^2 + bias[c]   (self loop contribution + bias)
__global__ void k_self_bias(const float* __restrict__ T, const float* __restrict__ dis,
                            const float* __restrict__ bias, float* __restrict__ out, int n) {
    int t = blockIdx.x * blockDim.x + threadIdx.x;
    if (t >= n * HIDDEN) return;
    int i = t >> 6, c = t & 63;
    float d = dis[i];
    out[t] = T[t] * d * d + bias[c];
}

// scatter-add: out[col[e]][:] += T[row[e]][:] * norm[e]  (16 threads/edge, 4 channels each)
__global__ void k_edge_scatter(const int* __restrict__ row, const int* __restrict__ col,
                               const float* __restrict__ norm, const float* __restrict__ T,
                               float* __restrict__ out, int e) {
    int t = blockIdx.x * blockDim.x + threadIdx.x;
    int ei = t >> 4;
    if (ei >= e) return;
    int g = (t & 15) * 4;
    int r = row[ei], c = col[ei];
    float w = norm[ei];
    const float4 v = *(const float4*)(T + (size_t)r * HIDDEN + g);
    float* dst = out + (size_t)c * HIDDEN + g;
    atomicAdd(dst + 0, v.x * w);
    atomicAdd(dst + 1, v.y * w);
    atomicAdd(dst + 2, v.z * w);
    atomicAdd(dst + 3, v.w * w);
}

__global__ void k_relu(float* h, int n) {
    int t = blockIdx.x * blockDim.x + threadIdx.x;
    if (t < n) h[t] = fmaxf(h[t], 0.0f);
}

// ---------------- pooling + final linear ----------------

__global__ void k_pool_zero(float* sums, float* cnt) {
    int t = blockIdx.x * blockDim.x + threadIdx.x;
    if (t < N_GRAPHS * HIDDEN) sums[t] = 0.0f;
    if (t < N_GRAPHS) cnt[t] = 0.0f;
}

__global__ void k_pool_accum(const float* __restrict__ h, const int* __restrict__ batch,
                             float* __restrict__ sums, float* __restrict__ cnt, int n) {
    int t = blockIdx.x * blockDim.x + threadIdx.x;
    int i = t >> 4;
    if (i >= n) return;
    int g = (t & 15) * 4;
    int b = batch[i];
    const float4 v = *(const float4*)(h + (size_t)i * HIDDEN + g);
    float* d = sums + (size_t)b * HIDDEN + g;
    atomicAdd(d + 0, v.x);
    atomicAdd(d + 1, v.y);
    atomicAdd(d + 2, v.z);
    atomicAdd(d + 3, v.w);
    if ((t & 15) == 0) atomicAdd(&cnt[b], 1.0f);
}

__global__ void k_final(const float* __restrict__ sums, const float* __restrict__ cnt,
                        const float* __restrict__ Wlin, const float* __restrict__ blin,
                        float* __restrict__ out) {
    int t = blockIdx.x * blockDim.x + threadIdx.x;
    if (t >= N_GRAPHS * N_OUT) return;
    int g = t / N_OUT, o = t % N_OUT;
    float inv = 1.0f / fmaxf(cnt[g], 1.0f);
    float acc = blin[o];
    for (int k = 0; k < HIDDEN; ++k)
        acc += (sums[g * HIDDEN + k] * inv) * Wlin[k * N_OUT + o];
    out[t] = acc;
}

// ---------------- launch ----------------

extern "C" void kernel_launch(void* const* d_in, const int* in_sizes, int n_in,
                              void* d_out, int out_size, void* d_ws, size_t ws_size,
                              hipStream_t stream) {
    const float* x     = (const float*)d_in[0];
    const int*   ei    = (const int*)d_in[1];
    const int*   batch = (const int*)d_in[2];
    const float* W1    = (const float*)d_in[3];
    const float* b1    = (const float*)d_in[4];
    const float* W2    = (const float*)d_in[5];
    const float* b2    = (const float*)d_in[6];
    const float* W3    = (const float*)d_in[7];
    const float* b3    = (const float*)d_in[8];
    const float* Wlin  = (const float*)d_in[9];
    const float* blin  = (const float*)d_in[10];
    float* out = (float*)d_out;

    const int N = N_NODES, E = N_EDGES;
    const int* row = ei;
    const int* col = ei + E;

    // workspace carve-up (256B aligned slots)
    char* ws = (char*)d_ws;
    size_t off = 0;
    auto alloc = [&](size_t bytes) { void* p = ws + off; off += (bytes + 255) & ~(size_t)255; return p; };
    float*    dis  = (float*)alloc((size_t)N * 4);                 // degree -> rsqrt(degree)
    float*    nrm  = (float*)alloc((size_t)E * 4);
    float*    T    = (float*)alloc((size_t)N * HIDDEN * 4);        // GEMM output
    float*    H    = (float*)alloc((size_t)N * HIDDEN * 4);        // aggregated output
    _Float16* Wt   = (_Float16*)alloc((size_t)HIDDEN * N_FEAT * 2);
    float*    sums = (float*)alloc((size_t)N_GRAPHS * HIDDEN * 4);
    float*    cnt  = (float*)alloc((size_t)N_GRAPHS * 4);
    (void)ws_size; (void)n_in; (void)in_sizes; (void)out_size;

    const int B = 256;
    auto gridFor = [](long long n, int b) { return (unsigned)((n + b - 1) / b); };

    // ---- gcn_norm ----
    k_deg_init<<<gridFor(N, B), B, 0, stream>>>(dis, N);
    k_deg_accum<<<gridFor(E, B), B, 0, stream>>>(col, dis, E);
    k_rsqrt<<<gridFor(N, B), B, 0, stream>>>(dis, N);
    k_norm<<<gridFor(E, B), B, 0, stream>>>(row, col, dis, nrm, E);

    const unsigned gemmGrid = N / 32;                 // 100000 % 32 == 0
    const unsigned scatGrid = gridFor((long long)E * 16, B);
    const unsigned nhGrid   = gridFor((long long)N * HIDDEN, B);

    // ---- layer 1: K = 128 ----
    k_wt<<<gridFor(N_FEAT * HIDDEN, B), B, 0, stream>>>(W1, Wt, N_FEAT, HIDDEN);
    k_gemm_wmma<<<gemmGrid, B, 0, stream>>>(x, N_FEAT, Wt, T);
    k_self_bias<<<nhGrid, B, 0, stream>>>(T, dis, b1, H, N);
    k_edge_scatter<<<scatGrid, B, 0, stream>>>(row, col, nrm, T, H, E);
    k_relu<<<nhGrid, B, 0, stream>>>(H, N * HIDDEN);

    // ---- layer 2: K = 64 ----
    k_wt<<<gridFor(HIDDEN * HIDDEN, B), B, 0, stream>>>(W2, Wt, HIDDEN, HIDDEN);
    k_gemm_wmma<<<gemmGrid, B, 0, stream>>>(H, HIDDEN, Wt, T);
    k_self_bias<<<nhGrid, B, 0, stream>>>(T, dis, b2, H, N);
    k_edge_scatter<<<scatGrid, B, 0, stream>>>(row, col, nrm, T, H, E);
    k_relu<<<nhGrid, B, 0, stream>>>(H, N * HIDDEN);

    // ---- layer 3: K = 64, no relu ----
    k_wt<<<gridFor(HIDDEN * HIDDEN, B), B, 0, stream>>>(W3, Wt, HIDDEN, HIDDEN);
    k_gemm_wmma<<<gemmGrid, B, 0, stream>>>(H, HIDDEN, Wt, T);
    k_self_bias<<<nhGrid, B, 0, stream>>>(T, dis, b3, H, N);
    k_edge_scatter<<<scatGrid, B, 0, stream>>>(row, col, nrm, T, H, E);

    // ---- mean pool + linear ----
    k_pool_zero<<<gridFor(N_GRAPHS * HIDDEN, B), B, 0, stream>>>(sums, cnt);
    k_pool_accum<<<gridFor((long long)N * 16, B), B, 0, stream>>>(H, batch, sums, cnt, N);
    k_final<<<gridFor(N_GRAPHS * N_OUT, B), B, 0, stream>>>(sums, cnt, Wlin, blin, out);
}
